// AttnOnlyTransformer_55929064128766
// MI455X (gfx1250) — matrix operand order repaired and checked by reading, not compiled, not run on hardware
//
#include <hip/hip_runtime.h>
#include <hip/hip_bf16.h>

// ---------------------------------------------------------------------------
// AttnOnlyTransformer for MI455X (gfx1250): bf16 WMMA pipeline, round 2
//   B=4, S=2048, D=2048 (d_model == vocab), DQK=128, L=4
// All matmuls via V_WMMA_F32_16X16X32_BF16 (fp32 accumulate), fp32 softmax.
// GEMMs: register-resident 32x64 (or 16x64) wave tiles, software-pipelined
// direct global b128 fragment loads, no LDS serialization in the hot loop.
// ---------------------------------------------------------------------------

typedef __bf16 bf16;
typedef __attribute__((ext_vector_type(16))) __bf16 v16bf;
typedef __attribute__((ext_vector_type(8)))  __bf16 v8bf;
typedef __attribute__((ext_vector_type(8)))  float  v8f;

#define BB   4
#define SS   2048
#define DD   2048
#define DQK  128
#define LL   4

// ---------------------------------------------------------------------------
// WMMA fragment helpers (layouts per CDNA5 ISA 7.12.2, wave32)
// ---------------------------------------------------------------------------

// A fragment: 16x32 bf16 tile, row-major, leading dim ld (elements).
// lane: m = lane%16 (row), g = lane/16. elems 0..7 -> K=g*8+e ; 8..15 -> K=16+g*8+(e-8)
__device__ __forceinline__ v16bf load_a_frag(const bf16* __restrict__ base, int ld) {
  int lane = threadIdx.x & 31;
  int m = lane & 15;
  int g = lane >> 4;
  const bf16* p = base + (size_t)m * ld + g * 8;
  v8bf lo = *(const v8bf*)(p);        // K = g*8 .. g*8+7
  v8bf hi = *(const v8bf*)(p + 16);   // K = 16+g*8 .. 16+g*8+7
  return __builtin_shufflevector(lo, hi, 0,1,2,3,4,5,6,7,8,9,10,11,12,13,14,15);
}

// B fragment: 32(K) x 16(N) tile stored "transposed": element (K,N) at base[N*ld + K].
// lane: n = lane%16 (col), g = lane/16. elems 0..7 -> K=g*8+e ; 8..15 -> K=16+g*8+(e-8)
__device__ __forceinline__ v16bf load_bt_frag(const bf16* __restrict__ base, int ld) {
  int lane = threadIdx.x & 31;
  int n = lane & 15;
  int g = lane >> 4;
  const bf16* p = base + (size_t)n * ld + g * 8;
  v8bf lo = *(const v8bf*)(p);
  v8bf hi = *(const v8bf*)(p + 16);
  return __builtin_shufflevector(lo, hi, 0,1,2,3,4,5,6,7,8,9,10,11,12,13,14,15);
}

__device__ __forceinline__ v8f wmma_bf16(v16bf a, v16bf b, v8f c) {
  return __builtin_amdgcn_wmma_f32_16x16x32_bf16(false, a, false, b, (short)0, c,
                                                 false, false);
}

// ---------------------------------------------------------------------------
// Kernel 0: weights [L,D,DQK] f32  ->  transposed bf16 [L,DQK,D]
// ---------------------------------------------------------------------------
__global__ void __launch_bounds__(256)
prep_w_kernel(const float* __restrict__ W, bf16* __restrict__ WT) {
  size_t idx = (size_t)blockIdx.x * 256 + threadIdx.x;
  size_t total = (size_t)LL * DD * DQK;
  if (idx >= total) return;
  int n = (int)(idx % DQK);
  size_t t = idx / DQK;
  int d = (int)(t % DD);
  int l = (int)(t / DD);
  WT[((size_t)l * DQK + n) * DD + d] = (bf16)W[idx];
}

// ---------------------------------------------------------------------------
// Kernel 1: enc = one_hot(token) + sinusoidal PE   (fp32 [B,S,D])
// ---------------------------------------------------------------------------
__global__ void __launch_bounds__(256)
build_enc_kernel(const int* __restrict__ tok, float* __restrict__ E) {
  size_t idx = (size_t)blockIdx.x * 256 + threadIdx.x;
  size_t total = (size_t)BB * SS * DD;
  if (idx >= total) return;
  int d = (int)(idx % DD);
  size_t t = idx / DD;
  int s = (int)(t % SS);
  int b = (int)(t / SS);
  int de = d & ~1;
  float div = __expf(-(logf(10000.0f) / (float)DD) * (float)de);
  float ang = (float)s * div;
  float pe = (d & 1) ? __cosf(ang) : __sinf(ang);
  float oh = (tok[(size_t)b * SS + s] == d) ? 1.0f : 0.0f;
  E[idx] = oh + pe;
}

// ---------------------------------------------------------------------------
// Kernel 2: fp32 enc -> bf16 enc [B,S,D] and bf16 enc^T [B,D,S] (LDS tile xpose)
//   grid (D/32, S/32, B), block (32, 8)
// ---------------------------------------------------------------------------
__global__ void __launch_bounds__(256)
conv_kernel(const float* __restrict__ E, bf16* __restrict__ Eb, bf16* __restrict__ ET) {
  __shared__ bf16 tile[32][33];
  int b  = blockIdx.z;
  int d0 = blockIdx.x * 32;
  int s0 = blockIdx.y * 32;
  const float* src = E  + ((size_t)b * SS + s0) * DD + d0;
  bf16*        eb  = Eb + ((size_t)b * SS + s0) * DD + d0;
  for (int i = threadIdx.y; i < 32; i += 8) {
    bf16 h = (bf16)src[(size_t)i * DD + threadIdx.x];
    eb[(size_t)i * DD + threadIdx.x] = h;
    tile[i][threadIdx.x] = h;
  }
  __syncthreads();
  bf16* et = ET + ((size_t)b * DD + d0) * SS + s0;
  for (int i = threadIdx.y; i < 32; i += 8) {
    et[(size_t)i * SS + threadIdx.x] = tile[threadIdx.x][i];
  }
}

// ---------------------------------------------------------------------------
// Kernel 3: Q/K projection:  Out[BS, DQK] = Ebf[BS, D] @ WT^T   (WT: [DQK, D])
//   grid (DQK/64, BS/32), one wave per block; wave tile = 32 rows x 64 cols.
//   Software-pipelined: next K-step fragments load before current WMMAs.
// ---------------------------------------------------------------------------
__global__ void __launch_bounds__(32)
gemm_qk_kernel(const bf16* __restrict__ Ebf, const bf16* __restrict__ WT,
               bf16* __restrict__ Out) {
  int m0 = blockIdx.y * 32;
  int n0 = blockIdx.x * 64;
  const bf16* A0 = Ebf + (size_t)m0 * DD;
  const bf16* A1 = A0 + (size_t)16 * DD;
  const bf16* Bb = WT + (size_t)n0 * DD;

  v8f acc[8] = {};
  v16bf a0 = load_a_frag(A0, DD);
  v16bf a1 = load_a_frag(A1, DD);
  v16bf b0 = load_bt_frag(Bb + (size_t)0  * DD, DD);
  v16bf b1 = load_bt_frag(Bb + (size_t)16 * DD, DD);
  v16bf b2 = load_bt_frag(Bb + (size_t)32 * DD, DD);
  v16bf b3 = load_bt_frag(Bb + (size_t)48 * DD, DD);

  constexpr int KSTEPS = DD / 32;
  for (int ks = 0; ks < KSTEPS - 1; ++ks) {
    int kn = (ks + 1) * 32;
    __builtin_prefetch(A0 + kn + 32, 0, 1);
    __builtin_prefetch(Bb + kn + 32, 0, 1);
    v16bf na0 = load_a_frag(A0 + kn, DD);
    v16bf na1 = load_a_frag(A1 + kn, DD);
    v16bf nb0 = load_bt_frag(Bb + (size_t)0  * DD + kn, DD);
    v16bf nb1 = load_bt_frag(Bb + (size_t)16 * DD + kn, DD);
    v16bf nb2 = load_bt_frag(Bb + (size_t)32 * DD + kn, DD);
    v16bf nb3 = load_bt_frag(Bb + (size_t)48 * DD + kn, DD);
    acc[0] = wmma_bf16(a0, b0, acc[0]);
    acc[1] = wmma_bf16(a0, b1, acc[1]);
    acc[2] = wmma_bf16(a0, b2, acc[2]);
    acc[3] = wmma_bf16(a0, b3, acc[3]);
    acc[4] = wmma_bf16(a1, b0, acc[4]);
    acc[5] = wmma_bf16(a1, b1, acc[5]);
    acc[6] = wmma_bf16(a1, b2, acc[6]);
    acc[7] = wmma_bf16(a1, b3, acc[7]);
    a0 = na0; a1 = na1; b0 = nb0; b1 = nb1; b2 = nb2; b3 = nb3;
  }
  acc[0] = wmma_bf16(a0, b0, acc[0]);
  acc[1] = wmma_bf16(a0, b1, acc[1]);
  acc[2] = wmma_bf16(a0, b2, acc[2]);
  acc[3] = wmma_bf16(a0, b3, acc[3]);
  acc[4] = wmma_bf16(a1, b0, acc[4]);
  acc[5] = wmma_bf16(a1, b1, acc[5]);
  acc[6] = wmma_bf16(a1, b2, acc[6]);
  acc[7] = wmma_bf16(a1, b3, acc[7]);

  int lane = threadIdx.x & 31;
  int n = lane & 15, g = lane >> 4;
#pragma unroll
  for (int mt = 0; mt < 2; ++mt)
#pragma unroll
    for (int nt = 0; nt < 4; ++nt) {
      v8f c = acc[mt * 4 + nt];
      bf16* o = Out + (size_t)(m0 + mt * 16) * DQK + n0 + nt * 16 + n;
#pragma unroll
      for (int r = 0; r < 8; ++r)
        o[(size_t)(r + 8 * g) * DQK] = (bf16)c[r];
    }
}

// ---------------------------------------------------------------------------
// Kernel 4: scores[b,q,k] = (Q·K^T)/sqrt(dqk), causal mask -> -1e9  (fp32)
//   grid (S/64 kgroups, S/16 qtiles, B), one wave; tile = 16 q x 64 k.
//   Fully-masked k-groups skip all compute.
// ---------------------------------------------------------------------------
__global__ void __launch_bounds__(32)
scores_kernel(const bf16* __restrict__ Q, const bf16* __restrict__ K,
              float* __restrict__ Sout) {
  int b = blockIdx.z;
  int q0 = blockIdx.y * 16;
  int k0 = blockIdx.x * 64;
  int lane = threadIdx.x & 31;
  int n = lane & 15, g = lane >> 4;
  float* out = Sout + ((size_t)b * SS + q0) * SS + k0;
  if (k0 > q0 + 15) {  // uniform: whole 16x64 group above diagonal
#pragma unroll
    for (int nt = 0; nt < 4; ++nt)
#pragma unroll
      for (int r = 0; r < 8; ++r)
        out[(size_t)(r + 8 * g) * SS + nt * 16 + n] = -1e9f;
    return;
  }
  const bf16* A  = Q + ((size_t)b * SS + q0) * DQK;
  const bf16* Bb = K + ((size_t)b * SS + k0) * DQK;
  v8f acc[4] = {};
#pragma unroll
  for (int d0 = 0; d0 < DQK; d0 += 32) {
    v16bf a = load_a_frag(A + d0, DQK);
#pragma unroll
    for (int nt = 0; nt < 4; ++nt) {
      v16bf bb = load_bt_frag(Bb + (size_t)(nt * 16) * DQK + d0, DQK);
      acc[nt] = wmma_bf16(a, bb, acc[nt]);
    }
  }
  const float scale = 0.08838834764831845f;  // 1/sqrt(128)
#pragma unroll
  for (int nt = 0; nt < 4; ++nt)
#pragma unroll
    for (int r = 0; r < 8; ++r) {
      int q = q0 + r + 8 * g, k = k0 + nt * 16 + n;
      float v = acc[nt][r] * scale;
      out[(size_t)(r + 8 * g) * SS + nt * 16 + n] = (k > q) ? -1e9f : v;
    }
}

// ---------------------------------------------------------------------------
// Kernel 5: row softmax over S=2048, fp32 in -> bf16 probabilities out.
//   one 256-thread block per row (b*S+q), LDS tree reductions.
// ---------------------------------------------------------------------------
__global__ void __launch_bounds__(256)
softmax_kernel(const float* __restrict__ Sb, bf16* __restrict__ P) {
  __shared__ float red[256];
  size_t row = blockIdx.x;
  const float* x = Sb + row * SS;
  bf16* p = P + row * SS;
  int t = threadIdx.x;
  float m = -1e30f;
  for (int i = t; i < SS; i += 256) m = fmaxf(m, x[i]);
  red[t] = m;
  __syncthreads();
  for (int s2 = 128; s2 > 0; s2 >>= 1) {
    if (t < s2) red[t] = fmaxf(red[t], red[t + s2]);
    __syncthreads();
  }
  m = red[0];
  __syncthreads();
  float sum = 0.0f;
  for (int i = t; i < SS; i += 256) sum += __expf(x[i] - m);
  red[t] = sum;
  __syncthreads();
  for (int s2 = 128; s2 > 0; s2 >>= 1) {
    if (t < s2) red[t] += red[t + s2];
    __syncthreads();
  }
  float inv = 1.0f / red[0];
  for (int i = t; i < SS; i += 256) p[i] = (bf16)(__expf(x[i] - m) * inv);
}

// ---------------------------------------------------------------------------
// Kernel 6: enc_new[b,q,d] = sum_k P[b,q,k] * enc[b,k,d]  (dominant GEMM)
//   grid (D/64, S/32, B), one wave; wave tile = 32 q x 64 d.
//   A = P (row-major, ld=S); B = enc^T [B,D,S] (K-contiguous). Pipelined.
// ---------------------------------------------------------------------------
__global__ void __launch_bounds__(32)
gemm_av_kernel(const bf16* __restrict__ P, const bf16* __restrict__ ET,
               float* __restrict__ Out) {
  int b = blockIdx.z;
  int q0 = blockIdx.y * 32;
  int d0 = blockIdx.x * 64;
  const bf16* A0 = P + ((size_t)b * SS + q0) * SS;
  const bf16* A1 = A0 + (size_t)16 * SS;
  const bf16* Bb = ET + ((size_t)b * DD + d0) * SS;

  v8f acc[8] = {};
  v16bf a0 = load_a_frag(A0, SS);
  v16bf a1 = load_a_frag(A1, SS);
  v16bf b0 = load_bt_frag(Bb + (size_t)0  * SS, SS);
  v16bf b1 = load_bt_frag(Bb + (size_t)16 * SS, SS);
  v16bf b2 = load_bt_frag(Bb + (size_t)32 * SS, SS);
  v16bf b3 = load_bt_frag(Bb + (size_t)48 * SS, SS);

  constexpr int KSTEPS = SS / 32;
  for (int ks = 0; ks < KSTEPS - 1; ++ks) {
    int kn = (ks + 1) * 32;
    __builtin_prefetch(A0 + kn + 32, 0, 1);
    __builtin_prefetch(Bb + kn + 32, 0, 1);
    v16bf na0 = load_a_frag(A0 + kn, SS);
    v16bf na1 = load_a_frag(A1 + kn, SS);
    v16bf nb0 = load_bt_frag(Bb + (size_t)0  * SS + kn, SS);
    v16bf nb1 = load_bt_frag(Bb + (size_t)16 * SS + kn, SS);
    v16bf nb2 = load_bt_frag(Bb + (size_t)32 * SS + kn, SS);
    v16bf nb3 = load_bt_frag(Bb + (size_t)48 * SS + kn, SS);
    acc[0] = wmma_bf16(a0, b0, acc[0]);
    acc[1] = wmma_bf16(a0, b1, acc[1]);
    acc[2] = wmma_bf16(a0, b2, acc[2]);
    acc[3] = wmma_bf16(a0, b3, acc[3]);
    acc[4] = wmma_bf16(a1, b0, acc[4]);
    acc[5] = wmma_bf16(a1, b1, acc[5]);
    acc[6] = wmma_bf16(a1, b2, acc[6]);
    acc[7] = wmma_bf16(a1, b3, acc[7]);
    a0 = na0; a1 = na1; b0 = nb0; b1 = nb1; b2 = nb2; b3 = nb3;
  }
  acc[0] = wmma_bf16(a0, b0, acc[0]);
  acc[1] = wmma_bf16(a0, b1, acc[1]);
  acc[2] = wmma_bf16(a0, b2, acc[2]);
  acc[3] = wmma_bf16(a0, b3, acc[3]);
  acc[4] = wmma_bf16(a1, b0, acc[4]);
  acc[5] = wmma_bf16(a1, b1, acc[5]);
  acc[6] = wmma_bf16(a1, b2, acc[6]);
  acc[7] = wmma_bf16(a1, b3, acc[7]);

  int lane = threadIdx.x & 31;
  int n = lane & 15, g = lane >> 4;
#pragma unroll
  for (int mt = 0; mt < 2; ++mt)
#pragma unroll
    for (int nt = 0; nt < 4; ++nt) {
      v8f c = acc[mt * 4 + nt];
      float* o = Out + ((size_t)b * SS + q0 + mt * 16) * DD + d0 + nt * 16 + n;
#pragma unroll
      for (int r = 0; r < 8; ++r)
        o[(size_t)(r + 8 * g) * DD] = c[r];
    }
}

// ---------------------------------------------------------------------------
// Host orchestration
// ---------------------------------------------------------------------------
extern "C" void kernel_launch(void* const* d_in, const int* in_sizes, int n_in,
                              void* d_out, int out_size, void* d_ws, size_t ws_size,
                              hipStream_t stream) {
  (void)in_sizes; (void)n_in; (void)out_size; (void)ws_size;
  const int*   tok = (const int*)d_in[0];
  const float* Wq  = (const float*)d_in[1];
  const float* Wk  = (const float*)d_in[2];
  float* out = (float*)d_out;

  char* p = (char*)d_ws;
  const size_t BSD = (size_t)BB * SS * DD;      // 16,777,216
  const size_t BSS = (size_t)BB * SS * SS;      // 16,777,216
  const size_t LDN = (size_t)LL * DD * DQK;     // 1,048,576
  const size_t BSQ = (size_t)BB * SS * DQK;     // 1,048,576

  float* E   = (float*)p; p += BSD * 4;         // fp32 enc [B,S,D]
  bf16*  Eb  = (bf16*)p;  p += BSD * 2;         // bf16 enc
  bf16*  ET  = (bf16*)p;  p += BSD * 2;         // bf16 enc^T [B,D,S]
  bf16*  WqT = (bf16*)p;  p += LDN * 2;         // [L,DQK,D]
  bf16*  WkT = (bf16*)p;  p += LDN * 2;
  bf16*  Qb  = (bf16*)p;  p += BSQ * 2;         // [B,S,DQK]
  bf16*  Kb  = (bf16*)p;  p += BSQ * 2;
  float* Sb  = (float*)p; p += BSS * 4;         // scores fp32 [B,S,S]
  bf16*  Pb  = (bf16*)p;  p += BSS * 2;         // probs bf16

  prep_w_kernel<<<(int)((LDN + 255) / 256), 256, 0, stream>>>(Wq, WqT);
  prep_w_kernel<<<(int)((LDN + 255) / 256), 256, 0, stream>>>(Wk, WkT);
  build_enc_kernel<<<(int)((BSD + 255) / 256), 256, 0, stream>>>(tok, E);

  for (int l = 0; l < LL; ++l) {
    conv_kernel<<<dim3(DD / 32, SS / 32, BB), dim3(32, 8), 0, stream>>>(E, Eb, ET);
    gemm_qk_kernel<<<dim3(DQK / 64, (BB * SS) / 32), 32, 0, stream>>>(
        Eb, WqT + (size_t)l * DQK * DD, Qb);
    gemm_qk_kernel<<<dim3(DQK / 64, (BB * SS) / 32), 32, 0, stream>>>(
        Eb, WkT + (size_t)l * DQK * DD, Kb);
    scores_kernel<<<dim3(SS / 64, SS / 16, BB), 32, 0, stream>>>(Qb, Kb, Sb);
    softmax_kernel<<<BB * SS, 256, 0, stream>>>(Sb, Pb);
    float* dst = (l == LL - 1) ? out : E;
    gemm_av_kernel<<<dim3(DD / 64, SS / 32, BB), 32, 0, stream>>>(Pb, ET, dst);
  }
}
